// Phi3Attention_65687229825906
// MI455X (gfx1250) — matrix operand order repaired and checked by reading, not compiled, not run on hardware
//
#include <hip/hip_runtime.h>

#define DEV __device__ __forceinline__

typedef __bf16 bf16;
typedef __bf16 v16bf __attribute__((ext_vector_type(16)));
typedef float v8f __attribute__((ext_vector_type(8)));
typedef unsigned int v8u __attribute__((ext_vector_type(8)));

constexpr int B_ = 2, L_ = 2048, D_ = 3072, H_ = 32, HD = 96;
constexpr int OPS = 9216;          // q(3072) + k(3072) + v(3072)
constexpr int QPOS = 3072, KPOS = 6144;

template <bool V> struct BoolC { static constexpr bool value = V; };

// ---------------- CDNA5 async memory->LDS helpers ----------------

// GLOBAL_LOAD_ASYNC_TO_LDS_B128: vdst = LDS byte offset (per-lane VGPR),
// vaddr = 64-bit global address. Tracked by ASYNCcnt.
DEV void async_copy_b128(unsigned lds_byte_off, const void* gaddr) {
  asm volatile("global_load_async_to_lds_b128 %0, %1, off"
               :
               : "v"(lds_byte_off), "v"(gaddr)
               : "memory");
}

DEV void wait_async0() {
  asm volatile("s_wait_asynccnt 0x0" ::: "memory");
}

DEV void wait_ds0() {
  asm volatile("s_wait_dscnt 0x0" ::: "memory");
}

// LDS byte offset of a __shared__ object: generic address low 32 bits are the
// LDS offset (aperture bits live in [63:32]).
DEV unsigned lds_off(const void* p) { return (unsigned)(uintptr_t)p; }

// ---------------- WMMA fragment helpers (CDNA5 16-bit layouts) ----------------

// A-matrix 16x32 (MxK): lane m=lane&15 holds row m.
// lanes 0-15: K=0..15, lanes16-31: K=16..31 (VGPR v -> K=ko+2v, ko+2v+1)
DEV v16bf frag_a(const bf16* base, int stride, int lane) {
  const int m = lane & 15;
  const int ko = (lane & 16) ? 16 : 0;
  const bf16* rp = base + (size_t)m * stride;
  v8u u;
#pragma unroll
  for (int v = 0; v < 8; ++v) {
    int k = 2 * v + ko;
    u[v] = *(const unsigned int*)(rp + k);
  }
  return __builtin_bit_cast(v16bf, u);
}

// B-matrix 32x16 (KxN) where B[k][n] = row-major W[n][k] (K contiguous per n).
// lane n=lane&15 holds column n; lanes0-15: K=0..15, lanes16-31: K=16..31.
DEV v16bf frag_b(const bf16* base, int stride, int lane) {
  const int n = lane & 15;
  const int ko = (lane & 16) ? 16 : 0;
  const bf16* rp = base + (size_t)n * stride;
  v8u u;
#pragma unroll
  for (int v = 0; v < 8; ++v) {
    u[v] = *(const unsigned int*)(rp + ko + 2 * v);
  }
  return __builtin_bit_cast(v16bf, u);
}

// B-matrix 32x16 fragment of V stored key-pair interleaved:
// Vt[pair][d][2] with element (key,d) at ((key>>1)*96 + d)*2 + (key&1).
// VGPR v needs keys (ko+2v, ko+2v+1) at one column -> one aligned b32 load.
DEV v16bf frag_b_vpair(const bf16* base, int col, int lane) {
  const int n = lane & 15;
  const int kp0 = (lane & 16) ? 8 : 0;  // pair index base (keys 16.. for hi lanes)
  v8u u;
#pragma unroll
  for (int v = 0; v < 8; ++v) {
    u[v] = *(const unsigned int*)(base + ((size_t)(kp0 + v) * HD + col + n) * 2);
  }
  return __builtin_bit_cast(v16bf, u);
}

DEV v8f wmma_bf16(v16bf a, v16bf b, v8f c) {
  return __builtin_amdgcn_wmma_f32_16x16x32_bf16(false, a, false, b, (short)0, c,
                                                 false, false);
}

// ---------------- fp32 -> bf16 convert ----------------

__global__ void k_f32_to_bf16(const float* __restrict__ in, bf16* __restrict__ out,
                              long n) {
  long i = (long)blockIdx.x * blockDim.x + threadIdx.x;
  long stride = (long)gridDim.x * blockDim.x;
  for (; i < n; i += stride) out[i] = (bf16)in[i];
}

// ---------------- GEMM: C[M,N] = A[MxK] * W[N,K]^T ----------------
// 128x128 block tile, 8 waves as 4x2; async-to-LDS double-buffered k-loop.
// Each wave: 32 rows x 64 cols = 8 C tiles; B fragment reused across 2 M tiles.

template <typename OutT>
__global__ __launch_bounds__(256) void k_gemm(const bf16* __restrict__ A,
                                              const bf16* __restrict__ Bm,
                                              OutT* __restrict__ C, int M, int N,
                                              int K) {
  __shared__ bf16 As[2][128 * 32];
  __shared__ bf16 Bs[2][128 * 32];
  const int tid = threadIdx.x;
  const int wave = tid >> 5;
  const int lane = tid & 31;
  const int wm = wave >> 1;   // 0..3: row block of 32
  const int wn = wave & 1;    // 0..1: col block of 64
  const int nBase = blockIdx.x * 128;
  const int mBase = blockIdx.y * 128;

  v8f acc[2][4] = {};

  // Issue async stage of one 32-wide k-slab into buffer `buf`.
  auto stage = [&](int buf, int kb) {
#pragma unroll
    for (int i = 0; i < 2; ++i) {  // A tile: 128 rows x 32 k = 512 x 16B
      int idx = tid + i * 256;
      int row = idx >> 2, seg = idx & 3;
      async_copy_b128(lds_off(&As[buf][row * 32 + seg * 8]),
                      A + (size_t)(mBase + row) * K + kb + seg * 8);
    }
#pragma unroll
    for (int i = 0; i < 2; ++i) {  // B tile: 128 rows x 32 k = 512 x 16B
      int idx = tid + i * 256;
      int row = idx >> 2, seg = idx & 3;
      async_copy_b128(lds_off(&Bs[buf][row * 32 + seg * 8]),
                      Bm + (size_t)(nBase + row) * K + kb + seg * 8);
    }
  };

  stage(0, 0);
  int pb = 0;
  for (int kb = 0; kb < K; kb += 32, pb ^= 1) {
    wait_async0();        // own async copies for buffer pb have landed
    __syncthreads();      // everyone's copies landed; prior reads of pb^1 done
    if (kb + 32 < K) stage(pb ^ 1, kb + 32);
    if (kb + 64 < K) {    // L2 prefetch two slabs ahead (global_prefetch_b8)
      __builtin_prefetch(A + (size_t)(mBase + (tid >> 1)) * K + kb + 64, 0, 1);
      __builtin_prefetch(Bm + (size_t)(nBase + (tid >> 1)) * K + kb + 64, 0, 1);
    }
    v16bf af[2];
#pragma unroll
    for (int mi = 0; mi < 2; ++mi)
      af[mi] = frag_a(&As[pb][(wm * 32 + mi * 16) * 32], 32, lane);
#pragma unroll
    for (int t = 0; t < 4; ++t) {
      v16bf bfrag = frag_b(&Bs[pb][(wn * 64 + t * 16) * 32], 32, lane);
#pragma unroll
      for (int mi = 0; mi < 2; ++mi)
        acc[mi][t] = wmma_bf16(af[mi], bfrag, acc[mi][t]);
    }
  }

  // C tile layout: VGPR r -> row r (lanes 0-15) / row r+8 (lanes 16-31); N = lane&15
#pragma unroll
  for (int mi = 0; mi < 2; ++mi)
#pragma unroll
    for (int t = 0; t < 4; ++t)
#pragma unroll
      for (int r = 0; r < 8; ++r) {
        int row = mBase + wm * 32 + mi * 16 + r + ((lane & 16) ? 8 : 0);
        int col = nBase + wn * 64 + t * 16 + (lane & 15);
        C[(size_t)row * N + col] = (OutT)acc[mi][t][r];
      }
}

// ---------------- RoPE (SuScaled) + scatter; V goes out key-pair interleaved ----------------

__global__ void k_rope(const bf16* __restrict__ qkv, bf16* __restrict__ Q,
                       bf16* __restrict__ Kc, bf16* __restrict__ Vc) {
  long t = (long)blockIdx.x * blockDim.x + threadIdx.x;
  const long total = (long)B_ * L_ * H_ * 48;
  if (t >= total) return;
  int i = (int)(t % 48);
  int h = (int)((t / 48) % H_);
  int l = (int)((t / (48 * H_)) % L_);
  int b = (int)(t / ((long)48 * H_ * L_));
  const bf16* row = qkv + (size_t)(b * L_ + l) * OPS;
  // inv_freq = 10000^(-2i/96); SHORT_FACTOR == 1
  float inv_freq = __expf(-(2.0f * (float)i / 96.0f) * 9.210340372f);
  float fr = (float)l * inv_freq;
  const float SC = 1.1902380714238083f;  // sqrt(1 + ln(32)/ln(4096))
  float c = __cosf(fr) * SC, s = __sinf(fr) * SC;
  size_t obase = ((size_t)(b * H_ + h) * L_ + l) * HD;
  const float qs = 0.10206207261596575f;  // 96^-0.5 folded into Q
  {
    float x1 = (float)row[h * HD + i];
    float x2 = (float)row[h * HD + i + 48];
    Q[obase + i]      = (bf16)((x1 * c - x2 * s) * qs);
    Q[obase + i + 48] = (bf16)((x2 * c + x1 * s) * qs);
  }
  {
    float x1 = (float)row[QPOS + h * HD + i];
    float x2 = (float)row[QPOS + h * HD + i + 48];
    Kc[obase + i]      = (bf16)(x1 * c - x2 * s);
    Kc[obase + i + 48] = (bf16)(x2 * c + x1 * s);
  }
  // V: element (key=l, d) at head base + ((l>>1)*96 + d)*2 + (l&1)
  bf16* vrow = Vc + (size_t)(b * H_ + h) * L_ * HD +
               ((size_t)(l >> 1) * HD) * 2 + (l & 1);
  vrow[(size_t)i * 2]        = row[KPOS + h * HD + i];
  vrow[(size_t)(i + 48) * 2] = row[KPOS + h * HD + i + 48];
}

// ---------------- Flash attention: 1 wave = 16 query rows, online softmax ----------------

__global__ __launch_bounds__(256) void k_attn(const bf16* __restrict__ Q,
                                              const bf16* __restrict__ Kc,
                                              const bf16* __restrict__ Vc,
                                              bf16* __restrict__ O) {
  __shared__ bf16 Pst[8][16 * 32];  // per-wave P staging (C-layout -> A-layout)
  __shared__ bf16 Vst[8][32 * 96];  // per-wave async-staged V block (6KB each)
  const int tid = threadIdx.x, wave = tid >> 5, lane = tid & 31;
  const int bh = blockIdx.x;
  const int b = bh >> 5, h = bh & 31;
  const int q0 = (blockIdx.y * 8 + wave) * 16;
  const bf16* Qh = Q + (size_t)bh * L_ * HD;
  const bf16* Kh = Kc + (size_t)bh * L_ * HD;
  const bf16* Vh = Vc + (size_t)bh * L_ * HD;  // key-pair interleaved layout

  v16bf aQ[3];
#pragma unroll
  for (int ks = 0; ks < 3; ++ks)
    aQ[ks] = frag_a(Qh + (size_t)q0 * HD + ks * 32, HD, lane);

  v8f accO[6] = {};
  float mrow[8], lrow[8];
#pragma unroll
  for (int r = 0; r < 8; ++r) { mrow[r] = -1e30f; lrow[r] = 0.0f; }

  bf16* pst = &Pst[wave][0];
  const unsigned vbase = lds_off(&Vst[wave][0]);
  const int kend = q0 + 16;  // causal: keys <= q0+15
  for (int kb = 0; kb < kend; kb += 32) {
    // async-stage this wave's V block: keys kb..kb+31 are the contiguous
    // 6144 bytes starting at byte kb*192 in the pair-interleaved layout;
    // overlaps with the QK^T WMMAs below.
    {
      const char* vsrc = (const char*)Vh + (size_t)kb * 192;
#pragma unroll
      for (int j = 0; j < 12; ++j) {
        unsigned off = (unsigned)(lane + 32 * j) * 16u;
        async_copy_b128(vbase + off, vsrc + off);
      }
    }
    // S = Q K^T  (two 16x16 n-tiles over 32 keys, 3 k-steps over head_dim 96)
    v8f S0 = {}, S1 = {};
#pragma unroll
    for (int ks = 0; ks < 3; ++ks) {
      v16bf bK0 = frag_b(Kh + (size_t)kb * HD + ks * 32, HD, lane);
      S0 = wmma_bf16(aQ[ks], bK0, S0);
      v16bf bK1 = frag_b(Kh + (size_t)(kb + 16) * HD + ks * 32, HD, lane);
      S1 = wmma_bf16(aQ[ks], bK1, S1);
    }
    float alpha[8];
    // Online-softmax update; MASKED specialization only on the diagonal block.
    auto update = [&](auto mc) {
      constexpr bool MASKED = decltype(mc)::value;
#pragma unroll
      for (int r = 0; r < 8; ++r) {
        float s0 = S0[r], s1 = S1[r];
        if (MASKED) {
          const int qrow = q0 + r + ((lane & 16) ? 8 : 0);
          const int key = kb + (lane & 15);
          s0 = (key <= qrow) ? s0 : -1e30f;
          s1 = (key + 16 <= qrow) ? s1 : -1e30f;
        }
        float mx = fmaxf(s0, s1);
#pragma unroll
        for (int d = 1; d < 16; d <<= 1) mx = fmaxf(mx, __shfl_xor(mx, d, 32));
        float mnew = fmaxf(mrow[r], mx);
        float al = __expf(mrow[r] - mnew);
        float p0 = __expf(s0 - mnew);
        float p1 = __expf(s1 - mnew);
        float ls = p0 + p1;
#pragma unroll
        for (int d = 1; d < 16; d <<= 1) ls += __shfl_xor(ls, d, 32);
        lrow[r] = lrow[r] * al + ls;
        mrow[r] = mnew;
        alpha[r] = al;
        const int prow = r + ((lane & 16) ? 8 : 0);
        pst[prow * 32 + (lane & 15)] = (bf16)p0;
        pst[prow * 32 + 16 + (lane & 15)] = (bf16)p1;
      }
    };
    if (kb + 31 <= q0) update(BoolC<false>{});  // interior block: no mask
    else               update(BoolC<true>{});   // diagonal block
#pragma unroll
    for (int t = 0; t < 6; ++t)
#pragma unroll
      for (int r = 0; r < 8; ++r) accO[t][r] = accO[t][r] * alpha[r];
    wait_ds0();      // P staging visible (same-wave LDS RAW, split DS counter)
    wait_async0();   // V block landed in LDS
    v16bf aP = frag_a(pst, 32, lane);
#pragma unroll
    for (int t = 0; t < 6; ++t) {
      v16bf bV = frag_b_vpair(&Vst[wave][0], t * 16, lane);
      accO[t] = wmma_bf16(aP, bV, accO[t]);
    }
  }

  // write attention output as bf16 rows of [B*L, H*96]
#pragma unroll
  for (int r = 0; r < 8; ++r) {
    const int q = q0 + r + ((lane & 16) ? 8 : 0);
    float inv = 1.0f / lrow[r];
    bf16* orow = O + (size_t)(b * L_ + q) * D_ + h * HD;
#pragma unroll
    for (int t = 0; t < 6; ++t)
      orow[t * 16 + (lane & 15)] = (bf16)(accO[t][r] * inv);
  }
}

// ---------------- host launch ----------------

extern "C" void kernel_launch(void* const* d_in, const int* in_sizes, int n_in,
                              void* d_out, int out_size, void* d_ws, size_t ws_size,
                              hipStream_t stream) {
  const float* x = (const float*)d_in[0];
  const float* Wqkv = (const float*)d_in[1];
  const float* Wo = (const float*)d_in[2];
  float* out = (float*)d_out;

  char* ws = (char*)d_ws;
  size_t off = 0;
  auto alloc = [&](size_t bytes) {
    char* p = ws + off;
    off += (bytes + 255) & ~(size_t)255;
    return p;
  };
  bf16* x_bf    = (bf16*)alloc((size_t)B_ * L_ * D_ * 2);
  bf16* wqkv_bf = (bf16*)alloc((size_t)OPS * D_ * 2);
  bf16* wo_bf   = (bf16*)alloc((size_t)D_ * D_ * 2);
  bf16* qkv_bf  = (bf16*)alloc((size_t)B_ * L_ * OPS * 2);
  bf16* q_bf    = (bf16*)alloc((size_t)B_ * H_ * L_ * HD * 2);
  bf16* k_bf    = (bf16*)alloc((size_t)B_ * H_ * L_ * HD * 2);
  bf16* v_bf    = (bf16*)alloc((size_t)B_ * H_ * L_ * HD * 2);
  bf16* attn_bf = (bf16*)alloc((size_t)B_ * L_ * D_ * 2);

  k_f32_to_bf16<<<2048, 256, 0, stream>>>(x, x_bf, (long)B_ * L_ * D_);
  k_f32_to_bf16<<<2048, 256, 0, stream>>>(Wqkv, wqkv_bf, (long)OPS * D_);
  k_f32_to_bf16<<<2048, 256, 0, stream>>>(Wo, wo_bf, (long)D_ * D_);

  dim3 g1(OPS / 128, (B_ * L_) / 128);
  k_gemm<bf16><<<g1, 256, 0, stream>>>(x_bf, wqkv_bf, qkv_bf, B_ * L_, OPS, D_);

  long rt = (long)B_ * L_ * H_ * 48;
  k_rope<<<(unsigned)((rt + 255) / 256), 256, 0, stream>>>(qkv_bf, q_bf, k_bf, v_bf);

  dim3 g2(B_ * H_, L_ / 128);
  k_attn<<<g2, 256, 0, stream>>>(q_bf, k_bf, v_bf, attn_bf);

  dim3 g3(D_ / 128, (B_ * L_) / 128);
  k_gemm<float><<<g3, 256, 0, stream>>>(attn_bf, wo_bf, out, B_ * L_, D_, D_);
}
